// GCN_dcaug_68633577390653
// MI455X (gfx1250) — matrix operand order repaired and checked by reading, not compiled, not run on hardware
//
#include <hip/hip_runtime.h>
#include <hip/hip_bf16.h>

typedef __attribute__((ext_vector_type(16))) __bf16 v16bf;
typedef __attribute__((ext_vector_type(8)))  __bf16 v8bf;
typedef __attribute__((ext_vector_type(8)))  float  v8f;

// ---------------------------------------------------------------------------
// Degree kernels: deg[i] = 1 (self loop) + #edges with dst==i; dinv = rsqrt.
// ---------------------------------------------------------------------------
__global__ void gcn_deg_init(float* __restrict__ deg, int n) {
    int i = blockIdx.x * blockDim.x + threadIdx.x;
    if (i < n) deg[i] = 1.0f;                       // self-loop contribution
}

__global__ void gcn_deg_count(float* __restrict__ deg, const int* __restrict__ dst, int E) {
    int e = blockIdx.x * blockDim.x + threadIdx.x;
    if (e < E) atomicAdd(&deg[dst[e]], 1.0f);
}

__global__ void gcn_deg_finalize(float* __restrict__ dinv, int n) {
    int i = blockIdx.x * blockDim.x + threadIdx.x;
    if (i < n) dinv[i] = rsqrtf(dinv[i]);           // deg >= 1 always
}

// ---------------------------------------------------------------------------
// fp32 -> (bf16 hi, bf16 lo) split planes: a ~= hi + lo, |lo| <= 2^-8 |a|.
// ---------------------------------------------------------------------------
__global__ void gcn_split_bf16(const float* __restrict__ in,
                               __bf16* __restrict__ hi, __bf16* __restrict__ lo,
                               size_t n) {
    size_t i = (size_t)blockIdx.x * blockDim.x + threadIdx.x;
    if (i >= n) return;
    float a = in[i];
    __bf16 h = (__bf16)a;
    hi[i] = h;
    lo[i] = (__bf16)(a - (float)h);
}

// ---------------------------------------------------------------------------
// Repack W[K,N] fp32 into WMMA B-fragment order (bf16 hi/lo):
//   out[(((nt*kchunks + kc8)*32 + lane)*16) + e] = split(W[k*N + col])
//   with k = kc8*32 + e + 16*(lane>>4), col = nt*16 + (lane&15).
// A lane's whole 16-element fragment is then 32 contiguous bytes.
// ---------------------------------------------------------------------------
__global__ void gcn_repack_w(const float* __restrict__ W,
                             __bf16* __restrict__ whi, __bf16* __restrict__ wlo,
                             int K, int N) {
    int gid = blockIdx.x * blockDim.x + threadIdx.x;
    if (gid >= K * N) return;
    const int kchunks = K >> 5;
    int e     = gid & 15;
    int lane  = (gid >> 4) & 31;
    int chunk = gid >> 9;                 // nt*kchunks + kc8
    int kc    = (chunk % kchunks) << 5;
    int nt    = chunk / kchunks;
    int k     = kc + e + ((lane >> 4) << 4);
    int col   = (nt << 4) + (lane & 15);
    float v   = W[(size_t)k * N + col];
    __bf16 h  = (__bf16)v;
    whi[gid]  = h;
    wlo[gid]  = (__bf16)(v - (float)h);
}

// ---------------------------------------------------------------------------
// WMMA GEMM: C[M,N] = A[M,K] @ B[K,N]; A given as split bf16 planes
// (row-major), B pre-repacked into fragment order. bf16x3 split product
// (hi*hi + hi*lo + lo*hi, f32 accumulate) for ~fp32 accuracy.
// One wave computes a 16x64 C tile (4 accumulators); 12 WMMA per k-chunk.
// Requires M%16==0, K%32==0, N%64==0. A and B fragments both use contiguous k = 16*half + e order.
// ---------------------------------------------------------------------------
__global__ void gcn_gemm_wmma_split(const __bf16* __restrict__ Ahi,
                                    const __bf16* __restrict__ Alo,
                                    const __bf16* __restrict__ Bhi,
                                    const __bf16* __restrict__ Blo,
                                    float* __restrict__ C,
                                    int M, int K, int N) {
    const int wave = blockIdx.x * (blockDim.x >> 5) + (threadIdx.x >> 5);
    const int lane = threadIdx.x & 31;
    const int mtiles  = M >> 4;
    const int ngroups = N >> 6;                    // 4 n-tiles per wave
    if (wave >= mtiles * ngroups) return;          // wave-uniform exit
    const int mt = wave % mtiles;                  // adjacent waves share B
    const int ng = wave / mtiles;
    const int half   = lane >> 4;
    const int lane16 = lane & 15;
    const int mrow   = (mt << 4) + lane16;
    const int kchunks = K >> 5;

    const __bf16* __restrict__ arh = Ahi + (size_t)mrow * K;
    const __bf16* __restrict__ arl = Alo + (size_t)mrow * K;
    const size_t jstride = (size_t)kchunks << 9;   // kchunks*32*16 elems per n-tile

    v8f acc0 = {}, acc1 = {}, acc2 = {}, acc3 = {};

    for (int kc8 = 0; kc8 < kchunks; ++kc8) {
        const int kc = kc8 << 5;

        // ---- A fragment (16x32): lane row mrow; elements 0-15 at k=16*half+0..15
        // -> two 16B loads per plane.
        v8bf x0 = *(const v8bf*)(arh + kc + half * 16);
        v8bf x1 = *(const v8bf*)(arh + kc + half * 16 + 8);
        v8bf y0 = *(const v8bf*)(arl + kc + half * 16);
        v8bf y1 = *(const v8bf*)(arl + kc + half * 16 + 8);
        v16bf ahi, alo;
        #pragma unroll
        for (int e = 0; e < 8; ++e) {
            ahi[e] = x0[e]; ahi[e + 8] = x1[e];
            alo[e] = y0[e]; alo[e + 8] = y1[e];
        }
        // prefetch next A chunk (speculative; dropped if OOB)
        __builtin_prefetch(arh + kc + 32, 0, 1);

        // ---- 4 B fragments, each 32 contiguous bytes per lane (L2-resident)
        size_t fb = ((size_t)((ng * 4) * kchunks + kc8) * 32 + lane) * 16;
        {
            v16bf bh = *(const v16bf*)(Bhi + fb);
            v16bf bl = *(const v16bf*)(Blo + fb);
            acc0 = __builtin_amdgcn_wmma_f32_16x16x32_bf16(false, ahi, false, bh, (short)0, acc0, false, false);
            acc0 = __builtin_amdgcn_wmma_f32_16x16x32_bf16(false, ahi, false, bl, (short)0, acc0, false, false);
            acc0 = __builtin_amdgcn_wmma_f32_16x16x32_bf16(false, alo, false, bh, (short)0, acc0, false, false);
        }
        {
            v16bf bh = *(const v16bf*)(Bhi + fb + jstride);
            v16bf bl = *(const v16bf*)(Blo + fb + jstride);
            acc1 = __builtin_amdgcn_wmma_f32_16x16x32_bf16(false, ahi, false, bh, (short)0, acc1, false, false);
            acc1 = __builtin_amdgcn_wmma_f32_16x16x32_bf16(false, ahi, false, bl, (short)0, acc1, false, false);
            acc1 = __builtin_amdgcn_wmma_f32_16x16x32_bf16(false, alo, false, bh, (short)0, acc1, false, false);
        }
        {
            v16bf bh = *(const v16bf*)(Bhi + fb + 2 * jstride);
            v16bf bl = *(const v16bf*)(Blo + fb + 2 * jstride);
            acc2 = __builtin_amdgcn_wmma_f32_16x16x32_bf16(false, ahi, false, bh, (short)0, acc2, false, false);
            acc2 = __builtin_amdgcn_wmma_f32_16x16x32_bf16(false, ahi, false, bl, (short)0, acc2, false, false);
            acc2 = __builtin_amdgcn_wmma_f32_16x16x32_bf16(false, alo, false, bh, (short)0, acc2, false, false);
        }
        {
            v16bf bh = *(const v16bf*)(Bhi + fb + 3 * jstride);
            v16bf bl = *(const v16bf*)(Blo + fb + 3 * jstride);
            acc3 = __builtin_amdgcn_wmma_f32_16x16x32_bf16(false, ahi, false, bh, (short)0, acc3, false, false);
            acc3 = __builtin_amdgcn_wmma_f32_16x16x32_bf16(false, ahi, false, bl, (short)0, acc3, false, false);
            acc3 = __builtin_amdgcn_wmma_f32_16x16x32_bf16(false, alo, false, bh, (short)0, acc3, false, false);
        }
    }

    // ---- C/D store: VGPR v -> (row = 16*mt + v + 8*half, col = lane16)
    const int ncol0 = (ng << 6) + lane16;
    #pragma unroll
    for (int v = 0; v < 8; ++v) {
        size_t row = (size_t)((mt << 4) + v + half * 8);
        float* crow = C + row * N + ncol0;
        crow[0]  = acc0[v];
        crow[16] = acc1[v];
        crow[32] = acc2[v];
        crow[48] = acc3[v];
    }
}

// ---------------------------------------------------------------------------
// Zero fill (float4 wide)
// ---------------------------------------------------------------------------
__global__ void gcn_zero_f4(float4* __restrict__ p, size_t n4) {
    size_t i = (size_t)blockIdx.x * blockDim.x + threadIdx.x;
    size_t stride = (size_t)gridDim.x * blockDim.x;
    float4 z; z.x = z.y = z.z = z.w = 0.0f;
    for (; i < n4; i += stride) p[i] = z;
}

// ---------------------------------------------------------------------------
// Edge scatter: agg[dst] += h[src] * dinv[src]*dinv[dst]; e>=E are the
// implicit self-loops (src=dst=e-E). float4 per thread, f32 global atomics.
// f4shift = log2(F/4).
// ---------------------------------------------------------------------------
__global__ void gcn_scatter(const float* __restrict__ h,
                            const int* __restrict__ src,
                            const int* __restrict__ dst,
                            const float* __restrict__ dinv,
                            float* __restrict__ agg,
                            int E, int n, int F, int f4shift) {
    size_t idx = (size_t)blockIdx.x * blockDim.x + threadIdx.x;
    size_t total = (size_t)(E + n) << f4shift;
    if (idx >= total) return;
    int e = (int)(idx >> f4shift);
    int f = ((int)idx & ((1 << f4shift) - 1)) << 2;
    int s, d;
    if (e < E) { s = src[e]; d = dst[e]; }
    else       { s = d = e - E; }
    float norm = dinv[s] * dinv[d];
    const float4 hv = *(const float4*)&h[(size_t)s * F + f];
    float* out = &agg[(size_t)d * F + f];
    atomicAdd(out + 0, hv.x * norm);
    atomicAdd(out + 1, hv.y * norm);
    atomicAdd(out + 2, hv.z * norm);
    atomicAdd(out + 3, hv.w * norm);
}

// ---------------------------------------------------------------------------
// out[i,j] = relu(out[i,j] + b[j]) in place; F power of two.
// ---------------------------------------------------------------------------
__global__ void gcn_bias_relu(float* __restrict__ buf, const float* __restrict__ b,
                              size_t total, int fmask) {
    size_t i = (size_t)blockIdx.x * blockDim.x + threadIdx.x;
    if (i >= total) return;
    float v = buf[i] + b[(int)i & fmask];
    buf[i] = v > 0.0f ? v : 0.0f;
}

// ---------------------------------------------------------------------------
// Host launcher
// ---------------------------------------------------------------------------
static inline size_t align256(size_t x) { return (x + 255) & ~(size_t)255; }

extern "C" void kernel_launch(void* const* d_in, const int* in_sizes, int n_in,
                              void* d_out, int out_size, void* d_ws, size_t ws_size,
                              hipStream_t stream) {
    const float* x  = (const float*)d_in[0];
    const int*   ei = (const int*)d_in[1];
    const float* W1 = (const float*)d_in[2];
    const float* b1 = (const float*)d_in[3];
    const float* W2 = (const float*)d_in[4];
    const float* b2 = (const float*)d_in[5];

    const int E   = in_sizes[1] / 2;          // edge_index is [2, E]
    const int F1  = in_sizes[3];              // hidden width  (256)
    const int F2  = in_sizes[5];              // output width  (128)
    const int Cin = in_sizes[2] / F1;         // input width   (128)
    const int Nn  = in_sizes[0] / Cin;        // nodes         (50000)

    const int* src = ei;
    const int* dst = ei + E;

    // Workspace: [dinv][h f32][agg f32][Ahi bf16][Alo bf16][Whi bf16][Wlo bf16]
    char* w = (char*)d_ws;
    size_t off = 0;
    float*  dinv = (float*)(w + off);  off += align256((size_t)Nn * 4);
    float*  hbuf = (float*)(w + off);  off += align256((size_t)Nn * F1 * 4);
    float*  aggb = (float*)(w + off);  off += align256((size_t)Nn * F1 * 4);
    __bf16* axhi = (__bf16*)(w + off); off += align256((size_t)Nn * F1 * 2);
    __bf16* axlo = (__bf16*)(w + off); off += align256((size_t)Nn * F1 * 2);
    size_t wElems = (size_t)(Cin * F1 > F1 * F2 ? Cin * F1 : F1 * F2);
    __bf16* wfhi = (__bf16*)(w + off); off += align256(wElems * 2);
    __bf16* wflo = (__bf16*)(w + off); off += align256(wElems * 2);

    const int B = 256;

    // ---- degrees & symmetric normalization
    gcn_deg_init    <<<(Nn + B - 1) / B, B, 0, stream>>>(dinv, Nn);
    gcn_deg_count   <<<(E  + B - 1) / B, B, 0, stream>>>(dinv, dst, E);
    gcn_deg_finalize<<<(Nn + B - 1) / B, B, 0, stream>>>(dinv, Nn);

    // ================= layer 1: h = x @ W1 =================
    {
        size_t nA = (size_t)Nn * Cin;
        gcn_split_bf16<<<(unsigned)((nA + B - 1) / B), B, 0, stream>>>(x, axhi, axlo, nA);
        gcn_repack_w<<<(Cin * F1 + B - 1) / B, B, 0, stream>>>(W1, wfhi, wflo, Cin, F1);
        int waves  = (Nn >> 4) * (F1 >> 6);
        int blocks = (waves + (B >> 5) - 1) / (B >> 5);
        gcn_gemm_wmma_split<<<blocks, B, 0, stream>>>(axhi, axlo, wfhi, wflo, hbuf, Nn, Cin, F1);
    }
    {   // aggregate + bias + relu  (aggb becomes layer-2 input)
        size_t n4 = (size_t)Nn * F1 / 4;
        gcn_zero_f4<<<4096, B, 0, stream>>>((float4*)aggb, n4);
        int f4s = 31 - __builtin_clz((unsigned)(F1 >> 2));
        size_t total = (size_t)(E + Nn) << f4s;
        gcn_scatter<<<(unsigned)((total + B - 1) / B), B, 0, stream>>>(
            hbuf, src, dst, dinv, aggb, E, Nn, F1, f4s);
        size_t tot = (size_t)Nn * F1;
        gcn_bias_relu<<<(unsigned)((tot + B - 1) / B), B, 0, stream>>>(aggb, b1, tot, F1 - 1);
    }

    // ================= layer 2: h2 = relu1 @ W2 (reuse hbuf) =================
    {
        size_t nA = (size_t)Nn * F1;
        gcn_split_bf16<<<(unsigned)((nA + B - 1) / B), B, 0, stream>>>(aggb, axhi, axlo, nA);
        gcn_repack_w<<<(F1 * F2 + B - 1) / B, B, 0, stream>>>(W2, wfhi, wflo, F1, F2);
        int waves  = (Nn >> 4) * (F2 >> 6);
        int blocks = (waves + (B >> 5) - 1) / (B >> 5);
        gcn_gemm_wmma_split<<<blocks, B, 0, stream>>>(axhi, axlo, wfhi, wflo, hbuf, Nn, F1, F2);
    }
    {   // aggregate into d_out + bias + relu
        float* out = (float*)d_out;
        size_t n4 = (size_t)Nn * F2 / 4;
        gcn_zero_f4<<<4096, B, 0, stream>>>((float4*)out, n4);
        int f4s = 31 - __builtin_clz((unsigned)(F2 >> 2));
        size_t total = (size_t)(E + Nn) << f4s;
        gcn_scatter<<<(unsigned)((total + B - 1) / B), B, 0, stream>>>(
            hbuf, src, dst, dinv, out, E, Nn, F2, f4s);
        size_t tot = (size_t)Nn * F2;
        gcn_bias_relu<<<(unsigned)((tot + B - 1) / B), B, 0, stream>>>(out, b2, tot, F2 - 1);
    }
}
